// BiSRConv2d_Down_64372969833106
// MI455X (gfx1250) — 8-bit matrix operand order repaired and checked by reading, compiled, not run on hardware
//
#include <hip/hip_runtime.h>

typedef __attribute__((ext_vector_type(8))) int v8i;
typedef __attribute__((ext_vector_type(4))) int v4i;

constexpr int NB = 16;
constexpr int NC = 128;
constexpr int PH = 64, PW = 64;
constexpr int HP = 66, WP = 66;
constexpr int KT = 18;

constexpr size_t POOLED_ELEMS = (size_t)NB * NC * PH * PW;
constexpr size_t SA_BYTES     = (size_t)NB * HP * WP * NC;
constexpr size_t WIMG_INTS    = (size_t)2 * 8 * KT * 32 * 8;
constexpr size_t POOL_OFF  = 0;
constexpr size_t SA1_OFF   = POOL_OFF + POOLED_ELEMS * 4;
constexpr size_t SA2_OFF   = SA1_OFF + SA_BYTES;
constexpr size_t WIMG_OFF  = SA2_OFF + SA_BYTES;
constexpr size_t SCALE_OFF = WIMG_OFF + WIMG_INTS * 4;

__global__ void zero16_kernel(char* p, unsigned n16) {
  unsigned i = blockIdx.x * 256u + threadIdx.x;
  if (i < n16) {
    v4i z = {0, 0, 0, 0};
    ((v4i*)p)[i] = z;
  }
}

__global__ void pool_sign_kernel(const float* __restrict__ x,
                                 const float* __restrict__ mvk1, const float* __restrict__ mvb1,
                                 const float* __restrict__ mvk2, const float* __restrict__ mvb2,
                                 float* __restrict__ pooled,
                                 signed char* __restrict__ s1,
                                 signed char* __restrict__ s2) {
  unsigned idx = blockIdx.x * 256u + threadIdx.x;
  int w = idx & 63;
  int h = (idx >> 6) & 63;
  int c = (idx >> 12) & 127;
  int b = idx >> 19;
  const float* xb = x + ((((size_t)b * 128 + c) * 128 + 2 * h) * 128 + 2 * w);
  float2 r0 = *(const float2*)xb;
  float2 r1 = *(const float2*)(xb + 128);
  float avg = (r0.x + r0.y + r1.x + r1.y) * 0.25f;
  pooled[idx] = avg;
  size_t sp = (((size_t)b * HP + h + 1) * WP + (w + 1)) * 128 + c;
  float a1 = avg * mvk1[c] + mvb1[c];
  float a2 = avg * mvk2[c] + mvb2[c];
  s1[sp] = a1 > 0.f ? 1 : (a1 < 0.f ? -1 : 0);
  s2[sp] = a2 > 0.f ? 1 : (a2 < 0.f ? -1 : 0);
}

__global__ void wscale_kernel(const float* __restrict__ W1, const float* __restrict__ W2,
                              float* __restrict__ scale) {
  __shared__ float red[256];
  int blk = blockIdx.x;
  int branch = blk >> 7, o = blk & 127;
  const float* Wp = (branch ? W2 : W1) + (size_t)o * 1152;
  float s = 0.f;
  for (int e = threadIdx.x; e < 1152; e += 256) s += fabsf(Wp[e]);
  red[threadIdx.x] = s;
  __syncthreads();
  for (int st = 128; st > 0; st >>= 1) {
    if ((int)threadIdx.x < st) red[threadIdx.x] += red[threadIdx.x + st];
    __syncthreads();
  }
  if (threadIdx.x == 0) scale[blk] = red[0] * (1.0f / 1152.0f);
}

__global__ void wimg_kernel(const float* __restrict__ W1, const float* __restrict__ W2,
                            int* __restrict__ wimg) {
  unsigned idx = blockIdx.x * 256u + threadIdx.x;
  int v = idx & 7;
  int lane = (idx >> 3) & 31;
  int rest = idx >> 8;
  int kt = rest % 18; rest /= 18;
  int oTile = rest & 7;
  int branch = rest >> 3;
  const float* Wp = branch ? W2 : W1;
  int o = oTile * 16 + (lane & 15);
  int lhalf = lane >> 4;
  int p = kt >> 1;
  int cbase = (kt & 1) << 6;
  unsigned d = 0;
  #pragma unroll
  for (int b4 = 0; b4 < 4; ++b4) {
    int K = (v >> 2) * 32 + lhalf * 16 + (v & 3) * 4 + b4;
    int cc = cbase + K;
    float wv = Wp[((size_t)o * 128 + cc) * 9 + p];
    int sb = wv > 0.f ? 1 : (wv < 0.f ? -1 : 0);
    d |= ((unsigned)(sb & 0xFF)) << (8 * b4);
  }
  wimg[idx] = (int)d;
}

__global__ __launch_bounds__(128) void bisr_gemm_kernel(
    const signed char* __restrict__ s1, const signed char* __restrict__ s2,
    const int* __restrict__ wimg, const float* __restrict__ scale,
    const float* __restrict__ pooled,
    const float* __restrict__ pb0_1, const float* __restrict__ al_1, const float* __restrict__ pb1_1,
    const float* __restrict__ pb0_2, const float* __restrict__ al_2, const float* __restrict__ pb1_2,
    float* __restrict__ out) {
  __shared__ signed char smem[3 * WP * 128];
  const int tid = threadIdx.x;
  const int lane = tid & 31;
  const int wv = tid >> 5;
  int bid = blockIdx.x;
  const int branch = bid >> 10; bid &= 1023;
  const int b = bid >> 6;
  const int h = bid & 63;

  const signed char* sA = branch ? s2 : s1;

  {
    const signed char* src = sA + ((size_t)b * HP + h) * (WP * 128);
    constexpr int N16 = 3 * WP * 128 / 16;
    for (int i = tid; i < N16; i += 128) {
      unsigned lds = (unsigned)(size_t)(&smem[i * 16]);
      unsigned long long ga = (unsigned long long)(size_t)(src + i * 16);
      asm volatile("global_load_async_to_lds_b128 %0, %1, off"
                   :: "v"(lds), "v"(ga) : "memory");
    }
    asm volatile("s_wait_asynccnt 0" ::: "memory");
    __syncthreads();
  }

  const int lN = lane & 15;
  const int lH = lane >> 4;
  const int oT0 = wv * 2, oT1 = oT0 + 1;
  const int* wA0 = wimg + (size_t)((branch * 8 + oT0) * KT) * 256;
  const int* wA1 = wimg + (size_t)((branch * 8 + oT1) * KT) * 256;

  v8i z = {0, 0, 0, 0, 0, 0, 0, 0};
  v8i acc0[4] = {z, z, z, z};
  v8i acc1[4] = {z, z, z, z};

  for (int kt = 0; kt < KT; ++kt) {
    const int p = kt >> 1;
    const int kh = p / 3;
    const int kw = p - kh * 3;
    const int cb = (kt & 1) << 6;
    union { v8i v; v4i q[2]; } A0, A1;
    const v4i* a0p = (const v4i*)(wA0 + (kt * 32 + lane) * 8);
    const v4i* a1p = (const v4i*)(wA1 + (kt * 32 + lane) * 8);
    A0.q[0] = a0p[0]; A0.q[1] = a0p[1];
    A1.q[0] = a1p[0]; A1.q[1] = a1p[1];
    const signed char* rp = &smem[(kh * WP + kw) * 128 + cb + lH * 16];
    #pragma unroll
    for (int n = 0; n < 4; ++n) {
      const signed char* bp = rp + (n * 16 + lN) * 128;
      union { v8i v; v4i q[2]; } Bm;
      Bm.q[0] = *(const v4i*)bp;
      Bm.q[1] = *(const v4i*)(bp + 32);
      acc0[n] = __builtin_amdgcn_wmma_i32_16x16x64_iu8(
          true, A0.v, true, Bm.v, acc0[n], false, false);
      acc1[n] = __builtin_amdgcn_wmma_i32_16x16x64_iu8(
          true, A1.v, true, Bm.v, acc1[n], false, false);
    }
  }

  const float* pb0 = branch ? pb0_2 : pb0_1;
  const float* al  = branch ? al_2  : al_1;
  const float* pb1 = branch ? pb1_2 : pb1_1;
  #pragma unroll
  for (int half = 0; half < 2; ++half) {
    const int oT = half ? oT1 : oT0;
    v8i* acc = half ? acc1 : acc0;
    float sc[8], f0[8], fa[8], f1[8];
    #pragma unroll
    for (int r = 0; r < 8; ++r) {
      int o = oT * 16 + r + lH * 8;
      sc[r] = scale[branch * 128 + o];
      f0[r] = pb0[o];
      fa[r] = al[o];
      f1[r] = pb1[o];
    }
    #pragma unroll
    for (int n = 0; n < 4; ++n) {
      int wpix = n * 16 + lN;
      #pragma unroll
      for (int r = 0; r < 8; ++r) {
        int o = oT * 16 + r + lH * 8;
        float f = (float)acc[n][r] * sc[r] + f0[r];
        f = f >= 0.f ? f : fa[r] * f;
        f += f1[r];
        size_t pidx = (((size_t)b * 128 + o) * 64 + h) * 64 + wpix;
        f += pooled[pidx];
        out[(((size_t)b * 256 + branch * 128 + o) * 64 + h) * 64 + wpix] = f;
      }
    }
  }
}

extern "C" void kernel_launch(void* const* d_in, const int* in_sizes, int n_in,
                              void* d_out, int out_size, void* d_ws, size_t ws_size,
                              hipStream_t stream) {
  (void)in_sizes; (void)n_in; (void)out_size; (void)ws_size;
  const float* x     = (const float*)d_in[0];
  const float* W1    = (const float*)d_in[1];
  const float* mvk1  = (const float*)d_in[2];
  const float* mvb1  = (const float*)d_in[3];
  const float* pb0_1 = (const float*)d_in[5];
  const float* al_1  = (const float*)d_in[6];
  const float* pb1_1 = (const float*)d_in[7];
  const float* W2    = (const float*)d_in[8];
  const float* mvk2  = (const float*)d_in[9];
  const float* mvb2  = (const float*)d_in[10];
  const float* pb0_2 = (const float*)d_in[12];
  const float* al_2  = (const float*)d_in[13];
  const float* pb1_2 = (const float*)d_in[14];
  float* out = (float*)d_out;

  char* ws = (char*)d_ws;
  float*       pooled = (float*)(ws + POOL_OFF);
  signed char* sA1    = (signed char*)(ws + SA1_OFF);
  signed char* sA2    = (signed char*)(ws + SA2_OFF);
  int*         wimg   = (int*)(ws + WIMG_OFF);
  float*       scale  = (float*)(ws + SCALE_OFF);

  {
    unsigned n16 = (unsigned)((SA_BYTES * 2) / 16);
    zero16_kernel<<<dim3((n16 + 255) / 256), dim3(256), 0, stream>>>((char*)sA1, n16);
  }
  pool_sign_kernel<<<dim3((unsigned)(POOLED_ELEMS / 256)), dim3(256), 0, stream>>>(
      x, mvk1, mvb1, mvk2, mvb2, pooled, sA1, sA2);
  wscale_kernel<<<dim3(256), dim3(256), 0, stream>>>(W1, W2, scale);
  wimg_kernel<<<dim3((unsigned)(WIMG_INTS / 256)), dim3(256), 0, stream>>>(W1, W2, wimg);
  bisr_gemm_kernel<<<dim3(2048), dim3(128), 0, stream>>>(
      sA1, sA2, wimg, scale, pooled,
      pb0_1, al_1, pb1_1, pb0_2, al_2, pb1_2, out);
}
